// ChEBIGIN_18459769438810
// MI455X (gfx1250) — compile-verified, matrix operand order checked by reading, not run on hardware
//
#include <hip/hip_runtime.h>
#include <hip/hip_bf16.h>
#include <stdint.h>

#define N_NODES   200000
#define N_EDGES   400000
#define NUM_GRAPHS 8192
#define NODE_DIM  80
#define EDGE_DIM  6
#define H_DIM     128
#define H2        256
#define NCLS      500
#define NLAYERS   4

typedef __attribute__((ext_vector_type(16))) __bf16 v16bf;
typedef __attribute__((ext_vector_type(8)))  __bf16 v8bf;
typedef __attribute__((ext_vector_type(8)))  float  v8f;

__device__ __forceinline__ unsigned short f2bf(float f) {
  unsigned int u = __float_as_uint(f);
  unsigned int r = u + 0x7fffu + ((u >> 16) & 1u);   // round-to-nearest-even
  return (unsigned short)(r >> 16);
}

__device__ __forceinline__ unsigned int pack2bf(float a, float b) {
  return (unsigned int)f2bf(a) | ((unsigned int)f2bf(b) << 16);
}

// ---------------- utility kernels ----------------

__global__ void zero_u32(unsigned int* __restrict__ p, int n) {
  int i = blockIdx.x * blockDim.x + threadIdx.x;
  if (i < n) p[i] = 0u;
}

// W[K][N] fp32 -> Wt[Npad][Kpad] bf16 (zero padded), i.e. B pre-transposed for WMMA
__global__ void convert_weight(const float* __restrict__ W, unsigned short* __restrict__ Wt,
                               int K, int N, int Kpad, int Npad) {
  int i = blockIdx.x * blockDim.x + threadIdx.x;
  if (i >= Kpad * Npad) return;
  int n = i / Kpad, k = i % Kpad;
  float v = (k < K && n < N) ? W[(size_t)k * N + n] : 0.f;
  Wt[i] = f2bf(v);
}

__global__ void bn_finalize(const float* __restrict__ colsum, const float* __restrict__ colsumsq,
                            const float* __restrict__ g, const float* __restrict__ b,
                            float* __restrict__ scale, float* __restrict__ shift,
                            int C, float invM) {
  int c = blockIdx.x * blockDim.x + threadIdx.x;
  if (c >= C) return;
  float m = colsum[c] * invM;
  float v = colsumsq[c] * invM - m * m;
  float sc = g[c] * rsqrtf(fmaxf(v, 0.f) + 1e-5f);
  scale[c] = sc;
  shift[c] = b[c] - m * sc;
}

__global__ void affine_relu_inplace(float* __restrict__ h, const float* __restrict__ scale,
                                    const float* __restrict__ shift, int n, int C) {
  int i = blockIdx.x * blockDim.x + threadIdx.x;
  if (i >= n) return;
  int c = i % C;
  h[i] = fmaxf(h[i] * scale[c] + shift[c], 0.f);
}

// ---------------- graph message passing ----------------
// agg[dst] += relu(h[src] + edge_attr @ edge_w + edge_b); edge embed recomputed (K=6, compute-free)
__global__ void edge_scatter(const float* __restrict__ h, const int* __restrict__ ei,
                             const float* __restrict__ edge_attr,
                             const float* __restrict__ edge_w, const float* __restrict__ edge_b,
                             float* __restrict__ agg) {
  int idx = blockIdx.x * blockDim.x + threadIdx.x;   // N_EDGES * 32 threads, 4 cols each
  if (idx >= N_EDGES * 32) return;
  int e  = idx >> 5;
  int c0 = (idx & 31) << 2;
  int s = ei[e];
  int d = ei[N_EDGES + e];
  float a[EDGE_DIM];
#pragma unroll
  for (int k = 0; k < EDGE_DIM; ++k) a[k] = edge_attr[(size_t)e * EDGE_DIM + k];
  float4 hv = *reinterpret_cast<const float4*>(&h[(size_t)s * H_DIM + c0]);
  float hvv[4] = {hv.x, hv.y, hv.z, hv.w};
#pragma unroll
  for (int j = 0; j < 4; ++j) {
    int c = c0 + j;
    float v = edge_b[c];
#pragma unroll
    for (int k = 0; k < EDGE_DIM; ++k) v += a[k] * edge_w[k * H_DIM + c];
    v += hvv[j];
    v = fmaxf(v, 0.f);
    atomicAdd(&agg[(size_t)d * H_DIM + c], v);
  }
}

// ---------------- pooling ----------------
__global__ void pool_kernel(const float* __restrict__ h, const int* __restrict__ batch,
                            float* __restrict__ psum, unsigned int* __restrict__ pmax,
                            float* __restrict__ counts) {
  int idx = blockIdx.x * blockDim.x + threadIdx.x;   // N_NODES * 32
  if (idx >= N_NODES * 32) return;
  int nd = idx >> 5;
  int c0 = (idx & 31) << 2;
  int g = batch[nd];
  float4 hv = *reinterpret_cast<const float4*>(&h[(size_t)nd * H_DIM + c0]);
  float vv[4] = {hv.x, hv.y, hv.z, hv.w};
#pragma unroll
  for (int j = 0; j < 4; ++j) {
    atomicAdd(&psum[(size_t)g * H_DIM + c0 + j], vv[j]);
    // post-ReLU values are >= 0 so uint bit pattern is order-preserving
    atomicMax(&pmax[(size_t)g * H_DIM + c0 + j], __float_as_uint(fmaxf(vv[j], 0.f)));
  }
  if (c0 == 0) atomicAdd(&counts[g], 1.f);
}

__global__ void build_p(const float* __restrict__ psum, const unsigned int* __restrict__ pmax,
                        const float* __restrict__ counts, float* __restrict__ p) {
  int idx = blockIdx.x * blockDim.x + threadIdx.x;
  if (idx >= NUM_GRAPHS * H2) return;
  int g = idx / H2, c = idx % H2;
  float v;
  if (c < H_DIM) v = psum[(size_t)g * H_DIM + c] / fmaxf(counts[g], 1.f);
  else           v = __uint_as_float(pmax[(size_t)g * H_DIM + (c - H_DIM)]);
  p[idx] = v;
}

// ---------------- fused WMMA bf16 GEMM ----------------
// C[M,Ntot] = epilogue( prologue(A0 (+A1)) @ Wt^T + bias )
// Template-specialized so the staging loop is branch-free on the fast path:
//   fast path = full 16-wide K chunk, row in bounds -> 4x float4 loads + 8x b32 LDS stores.
// Wt is bf16 [Npad][Kpad] (column n's K-vector contiguous) so B fragments are two b128 global loads.
#define BM 128
#define BN 64
#define KS 32
#define ASTRIDE 56   // halves; 112B row pitch = 7*16B -> 16B aligned, conflict-free half-wave reads

template<bool HAS_A1, bool HAS_AFF, bool PRO_RELU, bool HAS_STATS, bool EPI_RELU>
__global__ __launch_bounds__(256) void gemm_bf16(
    const float* __restrict__ A0, const float* __restrict__ A1,
    const float* __restrict__ pscale, const float* __restrict__ pshift,
    const unsigned short* __restrict__ Wt, const float* __restrict__ bias,
    float* __restrict__ C, float* __restrict__ colsum, float* __restrict__ colsumsq,
    int M, int K, int Kpad, int Ntot)
{
  __shared__ unsigned short As[BM * ASTRIDE];
  const int tid = threadIdx.x;
  const int blockM = blockIdx.x * BM;
  const int blockN = blockIdx.y * BN;
  const int w    = tid >> 5;
  const int lane = tid & 31;
  const int lm   = lane & 15;
  const int kb   = (lane >> 4) << 3;     // K sub-offset AND M sub-offset for C layout

  // staging coordinates (2 threads per row, 16 K-elements each)
  const int r   = tid >> 1;
  const int kl0 = (tid & 1) << 4;
  const int gm  = blockM + r;
  const float* arow  = A0 + (size_t)gm * K;
  const float* arow1 = HAS_A1 ? (A1 + (size_t)gm * K) : nullptr;
  unsigned int* lds_row = reinterpret_cast<unsigned int*>(&As[r * ASTRIDE + kl0]);

  v8f acc[4] = {};

  for (int k0 = 0; k0 < Kpad; k0 += KS) {
    __syncthreads();
    int gk0 = k0 + kl0;
    if (gm < M && gk0 + 16 <= K) {
      // fast path: vectorized, branch-free
      const float4* ap  = reinterpret_cast<const float4*>(arow + gk0);
      const float4* a1p = HAS_A1 ? reinterpret_cast<const float4*>(arow1 + gk0) : nullptr;
      const float4* scp = HAS_AFF ? reinterpret_cast<const float4*>(pscale + gk0) : nullptr;
      const float4* shp = HAS_AFF ? reinterpret_cast<const float4*>(pshift + gk0) : nullptr;
#pragma unroll
      for (int j = 0; j < 4; ++j) {
        float4 va = ap[j];
        if (HAS_A1) {
          float4 vb = a1p[j];
          va.x += vb.x; va.y += vb.y; va.z += vb.z; va.w += vb.w;
        }
        if (HAS_AFF) {
          float4 sc = scp[j], sh = shp[j];
          va.x = va.x * sc.x + sh.x; va.y = va.y * sc.y + sh.y;
          va.z = va.z * sc.z + sh.z; va.w = va.w * sc.w + sh.w;
          if (PRO_RELU) {
            va.x = fmaxf(va.x, 0.f); va.y = fmaxf(va.y, 0.f);
            va.z = fmaxf(va.z, 0.f); va.w = fmaxf(va.w, 0.f);
          }
        }
        lds_row[2 * j]     = pack2bf(va.x, va.y);
        lds_row[2 * j + 1] = pack2bf(va.z, va.w);
      }
    } else {
      // slow path: M-tail rows / fully out-of-range K chunks (zero fill)
#pragma unroll
      for (int j = 0; j < 8; ++j) {
        int gk = gk0 + 2 * j;
        float v0 = 0.f, v1 = 0.f;
        if (gm < M) {
          if (gk < K) {
            v0 = arow[gk];
            if (HAS_A1) v0 += arow1[gk];
            if (HAS_AFF) { v0 = v0 * pscale[gk] + pshift[gk]; if (PRO_RELU) v0 = fmaxf(v0, 0.f); }
          }
          if (gk + 1 < K) {
            v1 = arow[gk + 1];
            if (HAS_A1) v1 += arow1[gk + 1];
            if (HAS_AFF) { v1 = v1 * pscale[gk + 1] + pshift[gk + 1]; if (PRO_RELU) v1 = fmaxf(v1, 0.f); }
          }
        }
        lds_row[j] = pack2bf(v0, v1);
      }
    }
    __syncthreads();

    // A fragment: lane holds row M=w*16+lm; K = kb..kb+7 then 16+kb..16+kb+7
    int abase = (w * 16 + lm) * ASTRIDE;
    v8bf alo = *reinterpret_cast<const v8bf*>(&As[abase + kb]);
    v8bf ahi = *reinterpret_cast<const v8bf*>(&As[abase + 16 + kb]);
    v16bf a;
#pragma unroll
    for (int i = 0; i < 8; ++i) { a[i] = alo[i]; a[i + 8] = ahi[i]; }

#pragma unroll
    for (int nt = 0; nt < 4; ++nt) {
      int n = blockN + nt * 16 + lm;
      const unsigned short* bp = Wt + (size_t)n * Kpad + k0;
      v8bf blo = *reinterpret_cast<const v8bf*>(bp + kb);
      v8bf bhi = *reinterpret_cast<const v8bf*>(bp + 16 + kb);
      v16bf bb;
#pragma unroll
      for (int i = 0; i < 8; ++i) { bb[i] = blo[i]; bb[i + 8] = bhi[i]; }
      acc[nt] = __builtin_amdgcn_wmma_f32_16x16x32_bf16(
          false, a, false, bb, (short)0, acc[nt], false, false);
    }
  }

  // epilogue: C layout -> lane holds col n, rows mb..mb+7
  int mb = blockM + w * 16 + kb;
#pragma unroll
  for (int nt = 0; nt < 4; ++nt) {
    int n = blockN + nt * 16 + lm;
    float bv = (n < Ntot) ? bias[n] : 0.f;
    float s = 0.f, q = 0.f;
#pragma unroll
    for (int i = 0; i < 8; ++i) {
      float v = acc[nt][i] + bv;
      if (EPI_RELU) v = fmaxf(v, 0.f);
      int m = mb + i;
      if (m < M && n < Ntot) {
        C[(size_t)m * Ntot + n] = v;
        if (HAS_STATS) { s += v; q += v * v; }
      }
    }
    if (HAS_STATS) {
      s += __shfl_xor(s, 16, 32);
      q += __shfl_xor(q, 16, 32);
      if (lane < 16 && n < Ntot) {
        atomicAdd(&colsum[n], s);
        atomicAdd(&colsumsq[n], q);
      }
    }
  }
}

// ---------------- host-side orchestration ----------------

extern "C" void kernel_launch(void* const* d_in, const int* in_sizes, int n_in,
                              void* d_out, int out_size, void* d_ws, size_t ws_size,
                              hipStream_t stream) {
  (void)in_sizes; (void)n_in; (void)out_size; (void)ws_size;
  const float* x         = (const float*)d_in[0];
  const int*   ei        = (const int*)  d_in[1];
  const float* edge_attr = (const float*)d_in[2];
  const int*   batch     = (const int*)  d_in[3];
  const float* node_w    = (const float*)d_in[4];
  const float* node_b    = (const float*)d_in[5];
  const float* edge_w    = (const float*)d_in[6];
  const float* edge_b    = (const float*)d_in[7];
  const float* conv_w1   = (const float*)d_in[8];
  const float* conv_b1   = (const float*)d_in[9];
  const float* bn1g      = (const float*)d_in[10];
  const float* bn1b      = (const float*)d_in[11];
  const float* conv_w2   = (const float*)d_in[12];
  const float* conv_b2   = (const float*)d_in[13];
  const float* bn2g      = (const float*)d_in[14];
  const float* bn2b      = (const float*)d_in[15];
  const float* cls_w1    = (const float*)d_in[16];
  const float* cls_b1    = (const float*)d_in[17];
  const float* clsbng    = (const float*)d_in[18];
  const float* clsbnb    = (const float*)d_in[19];
  const float* cls_w2    = (const float*)d_in[20];
  const float* cls_b2    = (const float*)d_in[21];
  float* out = (float*)d_out;

  char* ws = (char*)d_ws;
  size_t off = 0;
  auto alloc = [&](size_t bytes) -> void* {
    off = (off + 255) & ~(size_t)255;
    void* p = ws + off;
    off += bytes;
    return p;
  };

  float* h   = (float*)alloc(sizeof(float) * (size_t)N_NODES * H_DIM);
  float* agg = (float*)alloc(sizeof(float) * (size_t)N_NODES * H_DIM);
  float* y   = (float*)alloc(sizeof(float) * (size_t)N_NODES * H2);

  float* poolbuf = (float*)alloc(sizeof(float) * (2 * (size_t)NUM_GRAPHS * H_DIM + NUM_GRAPHS));
  float* psum = poolbuf;
  unsigned int* pmax = (unsigned int*)(poolbuf + (size_t)NUM_GRAPHS * H_DIM);
  float* counts = poolbuf + 2 * (size_t)NUM_GRAPHS * H_DIM;
  float* p  = (float*)alloc(sizeof(float) * (size_t)NUM_GRAPHS * H2);
  float* zc = (float*)alloc(sizeof(float) * (size_t)NUM_GRAPHS * H_DIM);

  float* stats    = (float*)alloc(sizeof(float) * 2 * H2);  // colsum | colsumsq contiguous
  float* colsum   = stats;
  float* colsumsq = stats + H2;
  float* scalebuf = (float*)alloc(sizeof(float) * H2);
  float* shiftbuf = (float*)alloc(sizeof(float) * H2);

  unsigned short* wt_node = (unsigned short*)alloc(2 * (size_t)128 * 96);
  unsigned short* wt_c1   = (unsigned short*)alloc(2 * (size_t)NLAYERS * 256 * 128);
  unsigned short* wt_c2   = (unsigned short*)alloc(2 * (size_t)NLAYERS * 128 * 256);
  unsigned short* wt_cls1 = (unsigned short*)alloc(2 * (size_t)128 * 256);
  unsigned short* wt_cls2 = (unsigned short*)alloc(2 * (size_t)512 * 128);

  auto cdiv = [](int a, int b) { return (a + b - 1) / b; };

  // weight conversion (tiny; L2-resident afterwards)
  convert_weight<<<cdiv(128 * 96, 256), 256, 0, stream>>>(node_w, wt_node, NODE_DIM, H_DIM, 96, 128);
  for (int l = 0; l < NLAYERS; ++l) {
    convert_weight<<<cdiv(256 * 128, 256), 256, 0, stream>>>(
        conv_w1 + (size_t)l * H_DIM * H2, wt_c1 + (size_t)l * 256 * 128, H_DIM, H2, 128, 256);
    convert_weight<<<cdiv(128 * 256, 256), 256, 0, stream>>>(
        conv_w2 + (size_t)l * H2 * H_DIM, wt_c2 + (size_t)l * 128 * 256, H2, H_DIM, 256, 128);
  }
  convert_weight<<<cdiv(128 * 256, 256), 256, 0, stream>>>(cls_w1, wt_cls1, H2, H_DIM, 256, 128);
  convert_weight<<<cdiv(512 * 128, 256), 256, 0, stream>>>(cls_w2, wt_cls2, H_DIM, NCLS, 128, 512);

  // h = x @ node_w + node_b
  gemm_bf16<false, false, false, false, false>
      <<<dim3(cdiv(N_NODES, BM), 2), 256, 0, stream>>>(
      x, nullptr, nullptr, nullptr, wt_node, node_b, h, nullptr, nullptr,
      N_NODES, NODE_DIM, 96, H_DIM);

  for (int l = 0; l < NLAYERS; ++l) {
    zero_u32<<<cdiv(N_NODES * H_DIM, 256), 256, 0, stream>>>((unsigned int*)agg, N_NODES * H_DIM);
    edge_scatter<<<cdiv(N_EDGES * 32, 256), 256, 0, stream>>>(h, ei, edge_attr, edge_w, edge_b, agg);

    // y = (h + agg) @ w1 + b1   (+ column stats for BN1)
    zero_u32<<<2, 256, 0, stream>>>((unsigned int*)stats, 2 * H2);
    gemm_bf16<true, false, false, true, false>
        <<<dim3(cdiv(N_NODES, BM), 4), 256, 0, stream>>>(
        h, agg, nullptr, nullptr,
        wt_c1 + (size_t)l * 256 * 128, conv_b1 + (size_t)l * H2, y, colsum, colsumsq,
        N_NODES, H_DIM, 128, H2);
    bn_finalize<<<1, 256, 0, stream>>>(colsum, colsumsq, bn1g + (size_t)l * H2, bn1b + (size_t)l * H2,
                                       scalebuf, shiftbuf, H2, 1.f / N_NODES);

    // h = relu(bn1(y)) @ w2 + b2   (prologue applies bn1+relu; + column stats for BN2)
    zero_u32<<<2, 256, 0, stream>>>((unsigned int*)stats, 2 * H2);
    gemm_bf16<false, true, true, true, false>
        <<<dim3(cdiv(N_NODES, BM), 2), 256, 0, stream>>>(
        y, nullptr, scalebuf, shiftbuf,
        wt_c2 + (size_t)l * 128 * 256, conv_b2 + (size_t)l * H_DIM, h, colsum, colsumsq,
        N_NODES, H2, 256, H_DIM);
    bn_finalize<<<1, 256, 0, stream>>>(colsum, colsumsq, bn2g + (size_t)l * H_DIM, bn2b + (size_t)l * H_DIM,
                                       scalebuf, shiftbuf, H_DIM, 1.f / N_NODES);
    affine_relu_inplace<<<cdiv(N_NODES * H_DIM, 256), 256, 0, stream>>>(
        h, scalebuf, shiftbuf, N_NODES * H_DIM, H_DIM);
  }

  // global mean + max pooling
  int poolN = 2 * NUM_GRAPHS * H_DIM + NUM_GRAPHS;
  zero_u32<<<cdiv(poolN, 256), 256, 0, stream>>>((unsigned int*)poolbuf, poolN);
  pool_kernel<<<cdiv(N_NODES * 32, 256), 256, 0, stream>>>(h, batch, psum, pmax, counts);
  build_p<<<cdiv(NUM_GRAPHS * H2, 256), 256, 0, stream>>>(psum, pmax, counts, p);

  // classifier: z = relu(p @ w1 + b1); z = bn(z); out = z @ w2 + b2
  zero_u32<<<2, 256, 0, stream>>>((unsigned int*)stats, 2 * H2);
  gemm_bf16<false, false, false, true, true>
      <<<dim3(cdiv(NUM_GRAPHS, BM), 2), 256, 0, stream>>>(
      p, nullptr, nullptr, nullptr, wt_cls1, cls_b1, zc, colsum, colsumsq,
      NUM_GRAPHS, H2, 256, H_DIM);
  bn_finalize<<<1, 256, 0, stream>>>(colsum, colsumsq, clsbng, clsbnb,
                                     scalebuf, shiftbuf, H_DIM, 1.f / NUM_GRAPHS);
  gemm_bf16<false, true, false, false, false>
      <<<dim3(cdiv(NUM_GRAPHS, BM), 8), 256, 0, stream>>>(
      zc, nullptr, scalebuf, shiftbuf, wt_cls2, cls_b2, out, nullptr, nullptr,
      NUM_GRAPHS, H_DIM, 128, NCLS);
}
